// BertFusion_43404939493968
// MI455X (gfx1250) — matrix operand order repaired and checked by reading, not compiled, not run on hardware
//
#include <hip/hip_runtime.h>

// Fused BertFusion: O = softmax(H @ V^T) @ V, per batch.
// B=8, L=2048, V=1024, D=1024, fp32 in/out; bf16 WMMA with f32 accumulate.
//
// Pass 1 (pre-pass): convert V to bf16 once into d_ws, both row-major
//   (phase-1 B operand) and transposed [b][d][v] (phase-2 B operand).
//   Removes the per-WG v_cvt_pk_bf16_f32 flood and the phase-2 LDS
//   transpose staging + barriers from the main kernel.
// Pass 2 (main): per-WG 16-row stripe; S kept in accumulators, softmax via
//   lane shuffles + tiny LDS cross-wave reduction, 1/rowsum folded into
//   the final store.

typedef __attribute__((ext_vector_type(16))) __bf16 v16bf;
typedef __attribute__((ext_vector_type(8)))  __bf16 v8bf;
typedef __attribute__((ext_vector_type(4)))  __bf16 v4bf;
typedef __attribute__((ext_vector_type(8)))  float  v8f;

namespace {
constexpr int kB = 8;
constexpr int kL = 2048;
constexpr int kV = 1024;
constexpr int kD = 1024;
constexpr int BM = 16;                 // rows per workgroup
constexpr int NW = 8;                  // wave32s per workgroup
constexpr int STRIPE = kD / NW;        // 128 columns per wave
constexpr int NT = STRIPE / 16;        // 8 16x16 tiles per wave
constexpr int KC = 32;                 // WMMA K (bf16)

constexpr int HSP = 1032;              // padded pitch (bf16): conflict-free ds_load_b128

constexpr int LDS_HS_BYTES  = BM * HSP * 2;                 // 33024
constexpr int LDS_PS_OFF    = LDS_HS_BYTES;                 // 33024
constexpr int LDS_PS_BYTES  = BM * HSP * 2;                 // 33024
constexpr int LDS_RED_OFF   = LDS_PS_OFF + LDS_PS_BYTES;    // 66048
constexpr int LDS_RED_BYTES = 2 * NW * BM * 4;              // 1024
constexpr int LDS_TOTAL     = LDS_RED_OFF + LDS_RED_BYTES;  // 67072

constexpr size_t kVElems = (size_t)kB * kV * kD;            // 8M elements
} // namespace

__device__ __forceinline__ v16bf cat8(v8bf lo, v8bf hi) {
  return __builtin_shufflevector(lo, hi, 0,1,2,3,4,5,6,7,8,9,10,11,12,13,14,15);
}

// ---------------------------------------------------------------------------
// Pre-pass: V (f32) -> Vbf (bf16 row-major) + Vtbf (bf16 transposed [b][d][v])
// One 32x32 tile per 256-thread block, LDS tile for coalesced transpose.
// ---------------------------------------------------------------------------
__global__ __launch_bounds__(256)
void v_convert_kernel(const float* __restrict__ Vm,
                      __bf16* __restrict__ Vbf,
                      __bf16* __restrict__ Vtbf) {
  __shared__ __bf16 T[32][36];         // padded: conflict-free transpose reads
  const int tid = threadIdx.x;
  const int blk = blockIdx.x;
  constexpr int tilesD = kD / 32;              // 32
  constexpr int tilesPerBatch = (kV / 32) * tilesD;  // 1024
  const int b  = blk / tilesPerBatch;
  const int r  = blk % tilesPerBatch;
  const int v0 = (r / tilesD) * 32;
  const int d0 = (r % tilesD) * 32;

  const float* Vb = Vm   + (size_t)b * kV * kD;
  __bf16*      Rb = Vbf  + (size_t)b * kV * kD;
  __bf16*      Tb = Vtbf + (size_t)b * kD * kV;

  const int row  = tid >> 3;           // v within tile (0..31)
  const int col4 = (tid & 7) << 2;     // d within tile (0,4,..,28)

  const float4 q = *(const float4*)(Vb + (size_t)(v0 + row) * kD + d0 + col4);
  v4bf pk;
  pk[0] = (__bf16)q.x; pk[1] = (__bf16)q.y;
  pk[2] = (__bf16)q.z; pk[3] = (__bf16)q.w;
  *(v4bf*)(Rb + (size_t)(v0 + row) * kD + d0 + col4) = pk;
  T[row][col4 + 0] = pk[0]; T[row][col4 + 1] = pk[1];
  T[row][col4 + 2] = pk[2]; T[row][col4 + 3] = pk[3];
  __syncthreads();

  v4bf tp;                              // element (d0+row, v0+col4+j) = T[col4+j][row]
  tp[0] = T[col4 + 0][row]; tp[1] = T[col4 + 1][row];
  tp[2] = T[col4 + 2][row]; tp[3] = T[col4 + 3][row];
  *(v4bf*)(Tb + (size_t)(d0 + row) * kV + v0 + col4) = tp;
}

// ---------------------------------------------------------------------------
// Main fused kernel
// ---------------------------------------------------------------------------
__global__ __launch_bounds__(256)
void bert_fusion_kernel(const float* __restrict__ H,
                        const __bf16* __restrict__ Vbf,
                        const __bf16* __restrict__ Vtbf,
                        float* __restrict__ Out) {
  extern __shared__ char smem[];
  __bf16* Hs = (__bf16*)smem;                       // [16][HSP]
  __bf16* Ps = (__bf16*)(smem + LDS_PS_OFF);        // [16][HSP]
  float*  redmax = (float*)(smem + LDS_RED_OFF);            // [NW][16]
  float*  redsum = (float*)(smem + LDS_RED_OFF + NW*BM*4);  // [NW][16]

  const int tid  = threadIdx.x;
  const int wv   = tid >> 5;
  const int lane = tid & 31;
  const int lrow = lane & 15;   // tile row (A) or column (B/C)
  const int half = lane >> 4;   // lane half selects K / M split

  const int blk = blockIdx.x;
  const int b   = blk >> 7;           // / (L/16)
  const int m0  = (blk & 127) << 4;

  const float*  Hb  = H    + ((size_t)b * kL + m0) * kD;  // 16 x 1024
  const __bf16* Vrb = Vbf  + (size_t)b * kV * kD;         // 1024 x 1024 (row-major)
  const __bf16* Vtb = Vtbf + (size_t)b * kD * kV;         // 1024 x 1024 (d-major)
  float*        Ob  = Out  + ((size_t)b * kL + m0) * kD;

  const v8f vzero = {0.f,0.f,0.f,0.f,0.f,0.f,0.f,0.f};

  // ---------------- Stage H block (f32 -> bf16) into LDS ----------------
  {
    const float4* src = (const float4*)Hb;
    #pragma unroll
    for (int i = 0; i < 16; ++i) {
      const int idx = tid + i * 256;            // 4096 float4 chunks
      const float4 q = src[idx];
      const int row = idx >> 8;
      const int col = (idx & 255) << 2;
      __bf16* dst = Hs + row * HSP + col;
      dst[0] = (__bf16)q.x; dst[1] = (__bf16)q.y;
      dst[2] = (__bf16)q.z; dst[3] = (__bf16)q.w;
    }
  }
  __syncthreads();

  // ---------------- Phase 1: S = H @ V^T (per-wave 16x128 stripe) ----------------
  const int ncolBase = wv * STRIPE;
  v8f acc[NT];
  #pragma unroll
  for (int t = 0; t < NT; ++t) acc[t] = vzero;

  for (int kk = 0; kk < kD; kk += KC) {
    // A operand: H rows from LDS (ISA A-matrix lane layout)
    const v8bf a_lo = *(const v8bf*)(Hs + lrow * HSP + kk + half * 8);
    const v8bf a_hi = *(const v8bf*)(Hs + lrow * HSP + kk + 16 + half * 8);
    const v16bf a = cat8(a_lo, a_hi);
    #pragma unroll
    for (int t = 0; t < NT; ++t) {
      // B operand: bf16 V row streamed from L2, two 16-byte chunks (k = 8h.., 16+8h..)
      const int vrow = ncolBase + t * 16 + lrow;
      const __bf16* vp = Vrb + (size_t)vrow * kD + kk + half * 8;
      const v16bf bm = cat8(*(const v8bf*)vp, *(const v8bf*)(vp + 16));
      acc[t] = __builtin_amdgcn_wmma_f32_16x16x32_bf16(
          false, a, false, bm, (short)0, acc[t], false, false);
    }
  }

  // ---------------- Softmax over the V axis ----------------
  // C/D layout: vgpr r, lane -> row = r + 8*half, col = ncolBase + t*16 + lrow
  float lmax[8];
  #pragma unroll
  for (int r = 0; r < 8; ++r) {
    float m = acc[0][r];
    #pragma unroll
    for (int t = 1; t < NT; ++t) m = fmaxf(m, acc[t][r]);
    #pragma unroll
    for (int s = 1; s < 16; s <<= 1) m = fmaxf(m, __shfl_xor(m, s, 32));
    lmax[r] = m;
  }
  if (lrow == 0) {
    #pragma unroll
    for (int r = 0; r < 8; ++r) redmax[wv * BM + half * 8 + r] = lmax[r];
  }
  __syncthreads();

  float gmax[8];
  #pragma unroll
  for (int r = 0; r < 8; ++r) {
    float m = redmax[half * 8 + r];
    #pragma unroll
    for (int ww = 1; ww < NW; ++ww) m = fmaxf(m, redmax[ww * BM + half * 8 + r]);
    gmax[r] = m;
  }

  float lsum[8];
  #pragma unroll
  for (int r = 0; r < 8; ++r) {
    float s = 0.f;
    #pragma unroll
    for (int t = 0; t < NT; ++t) {
      const float e = __expf(acc[t][r] - gmax[r]);
      acc[t][r] = e;
      s += e;
    }
    #pragma unroll
    for (int sh = 1; sh < 16; sh <<= 1) s += __shfl_xor(s, sh, 32);
    lsum[r] = s;
  }
  if (lrow == 0) {
    #pragma unroll
    for (int r = 0; r < 8; ++r) redsum[wv * BM + half * 8 + r] = lsum[r];
  }
  // Write unnormalized P-hat (bf16) to LDS; 1/rowsum folded into final store.
  #pragma unroll
  for (int t = 0; t < NT; ++t)
    #pragma unroll
    for (int r = 0; r < 8; ++r)
      Ps[(half * 8 + r) * HSP + ncolBase + t * 16 + lrow] = (__bf16)acc[t][r];
  __syncthreads();

  float rinv[8];
  #pragma unroll
  for (int r = 0; r < 8; ++r) {
    float s = 0.f;
    #pragma unroll
    for (int ww = 0; ww < NW; ++ww) s += redsum[ww * BM + half * 8 + r];
    rinv[r] = 1.0f / s;
  }

  // ---------------- Phase 2: O = P-hat @ V (barrier-free) ----------------
  const int dBase = wv * STRIPE;
  v8f oacc[NT];
  #pragma unroll
  for (int t = 0; t < NT; ++t) oacc[t] = vzero;

  for (int kv = 0; kv < kV; kv += KC) {
    // A operand: P-hat rows from LDS
    const v8bf a_lo = *(const v8bf*)(Ps + lrow * HSP + kv + half * 8);
    const v8bf a_hi = *(const v8bf*)(Ps + lrow * HSP + kv + 16 + half * 8);
    const v16bf a = cat8(a_lo, a_hi);
    #pragma unroll
    for (int t = 0; t < NT; ++t) {
      // B operand: column d of V = contiguous row of transposed bf16 copy
      const int d = dBase + t * 16 + lrow;
      const __bf16* vp = Vtb + (size_t)d * kV + kv + half * 8;
      const v16bf bm = cat8(*(const v8bf*)vp, *(const v8bf*)(vp + 16));
      oacc[t] = __builtin_amdgcn_wmma_f32_16x16x32_bf16(
          false, a, false, bm, (short)0, oacc[t], false, false);
    }
  }

  // ---------------- Scale by 1/rowsum and store fp32 output ----------------
  #pragma unroll
  for (int t = 0; t < NT; ++t) {
    #pragma unroll
    for (int r = 0; r < 8; ++r) {
      const int row = half * 8 + r;
      const int d   = dBase + t * 16 + lrow;
      Ob[(size_t)row * kD + d] = oacc[t][r] * rinv[r];
    }
  }
}

extern "C" void kernel_launch(void* const* d_in, const int* in_sizes, int n_in,
                              void* d_out, int out_size, void* d_ws, size_t ws_size,
                              hipStream_t stream) {
  const float* H  = (const float*)d_in[0];   // (8, 2048, 1024) f32
  const float* Vm = (const float*)d_in[1];   // (8, 1024, 1024) f32
  float* Out = (float*)d_out;                // (8, 2048, 1024) f32

  // Workspace: bf16 V row-major (16 MB) + bf16 V transposed (16 MB)
  __bf16* Vbf  = (__bf16*)d_ws;
  __bf16* Vtbf = Vbf + kVElems;

  // Pre-pass: convert + transpose V once (stays L2-resident for main kernel)
  v_convert_kernel<<<dim3(kB * (kV / 32) * (kD / 32)), dim3(256), 0, stream>>>(
      Vm, Vbf, Vtbf);

  const dim3 grid(kB * (kL / BM));           // 1024 workgroups
  const dim3 block(NW * 32);                 // 256 threads = 8 wave32s
  bert_fusion_kernel<<<grid, block, LDS_TOTAL, stream>>>(H, Vbf, Vtbf, Out);
}
